// LSTMDecoderV1_35055523070268
// MI455X (gfx1250) — matrix operand order repaired and checked by reading, not compiled, not run on hardware
//
#include <hip/hip_runtime.h>

// ---------------------------------------------------------------------------
// 2-layer LSTM (B=256, T=512, C=H=256) for gfx1250 (MI455X).
// bf16 WMMA with f32 accumulation, weights resident in LDS (TDM-loaded),
// per-step h exchange between 8 slice-WGs per batch group via L2 + atomics.
// ---------------------------------------------------------------------------

#define B_   256
#define T_   512
#define C_   256
#define H_   256
#define NSLICE 8      // unit-dim split (32 units each -> 128 gate cols)
#define NGROUP 8      // batch split (32 rows each)

typedef __bf16 v16bf __attribute__((ext_vector_type(16)));
typedef float  v8f   __attribute__((ext_vector_type(8)));
typedef unsigned int u32x4v __attribute__((ext_vector_type(4)));
typedef int          i32x8v __attribute__((ext_vector_type(8)));
typedef int          i32x4v __attribute__((ext_vector_type(4)));

#ifndef USE_TDM
#if defined(__has_builtin)
#if __has_builtin(__builtin_amdgcn_tensor_load_to_lds) && __has_builtin(__builtin_amdgcn_s_wait_tensorcnt)
#define USE_TDM 1
#else
#define USE_TDM 0
#endif
#else
#define USE_TDM 0
#endif
#endif

__device__ __forceinline__ unsigned short f2bf(float f) {
  unsigned u = __float_as_uint(f);
  return (unsigned short)((u + 0x7FFFu + ((u >> 16) & 1u)) >> 16);  // RNE
}
__device__ __forceinline__ float bf2f(unsigned short h) {
  return __uint_as_float(((unsigned)h) << 16);
}
__device__ __forceinline__ float sigm(float x) { return 1.0f / (1.0f + __expf(-x)); }
__device__ __forceinline__ float tanh_fast(float x) {
  float e = __expf(-2.0f * x);
  return (1.0f - e) / (1.0f + e);
}

// ---------------------------------------------------------------------------
// Pack Wcat = [Wih ; Whh] (f32 [1024,256] each) into bf16 WMMA B-fragment
// order.  Packed column p = s*128 + gate*32 + u  (slice-major), source row
// orig = gate*256 + s*32 + u.  Fragment (ntg, kt) is 1 KB: lane*32 + v*4 + h*2
// with B layout: lanes 0-15 hold K%32 in 0-7,16-23, lanes 16-31 hold 8-15,24-31 (v=(K%32/16)*4+(K%8)/2).
// ---------------------------------------------------------------------------
__global__ void lstm_pack_weights(const float* __restrict__ Wih0,
                                  const float* __restrict__ Whh0,
                                  const float* __restrict__ Wih1,
                                  const float* __restrict__ Whh1,
                                  unsigned short* __restrict__ W0p,
                                  unsigned short* __restrict__ W1p) {
  unsigned idx = blockIdx.x * blockDim.x + threadIdx.x;   // 2 * 512 * 1024
  unsigned layer = idx >> 19;
  unsigned e = idx & 0x7FFFFu;
  unsigned k = e >> 10;          // 0..511  (0..255 = input part, 256.. = h part)
  unsigned p = e & 1023u;        // packed gate column
  unsigned s = p >> 7, r2 = p & 127u, gidx = r2 >> 5, u = r2 & 31u;
  unsigned orig = gidx * 256u + s * 32u + u;
  const float* Wih = layer ? Wih1 : Wih0;
  const float* Whh = layer ? Whh1 : Whh0;
  float val = (k < 256u) ? Wih[orig * 256u + k] : Whh[orig * 256u + (k - 256u)];
  unsigned kt = k >> 5, kk = k & 31u;
  unsigned lane = (p & 15u) + 16u * ((kk >> 3) & 1u);
  unsigned v = ((kk >> 4) << 2) + ((kk & 7u) >> 1);
  unsigned hh = kk & 1u;
  unsigned ntg = p >> 4;
  unsigned off = ((ntg * 16u + kt) << 10) + lane * 32u + v * 4u + hh * 2u;  // bytes
  (layer ? W1p : W0p)[off >> 1] = f2bf(val);
}

// ---------------------------------------------------------------------------
// Init: zero sync flags, broadcast h0 into exchange slot 0, pack biases into
// the slice-major column order used by the packed weights.
// ---------------------------------------------------------------------------
__global__ void lstm_init(const float* __restrict__ h0,
                          const float* __restrict__ bih0, const float* __restrict__ bhh0,
                          const float* __restrict__ bih1, const float* __restrict__ bhh1,
                          unsigned int* __restrict__ flags,
                          unsigned short* __restrict__ hb0,
                          unsigned short* __restrict__ hb1,
                          float* __restrict__ b0p, float* __restrict__ b1p) {
  unsigned idx = blockIdx.x * blockDim.x + threadIdx.x;   // 131072
  if (idx < 2u * NGROUP * T_) flags[idx] = 0u;
  if (idx < (unsigned)(B_ * H_)) {
    unsigned j = idx & 255u;
    hb0[idx] = f2bf(h0[j]);            // layer 0 initial h, broadcast over batch
    hb1[idx] = f2bf(h0[256u + j]);     // layer 1 initial h
  }
  if (idx < 2048u) {
    unsigned layer = idx >> 10, p = idx & 1023u;
    unsigned s = p >> 7, r2 = p & 127u, gidx = r2 >> 5, u = r2 & 31u;
    unsigned orig = gidx * 256u + s * 32u + u;
    float v = layer ? (bih1[orig] + bhh1[orig]) : (bih0[orig] + bhh0[orig]);
    (layer ? b1p : b0p)[p] = v;
  }
}

// ---------------------------------------------------------------------------
// Recurrent phase kernel: 64 WGs = 8 batch groups x 8 unit slices.
// LDS: 128 KB resident weight slice + 32 KB A-staging (aliased as gates).
// ---------------------------------------------------------------------------
__global__ __launch_bounds__(256, 1)
void lstm_phase(const float* __restrict__ xf,          // phase0: inputs [B,T,C] f32
                const unsigned short* __restrict__ xb, // phase1: layer0 output hb0
                int is_f32,
                const unsigned short* __restrict__ Wp, // frag-packed weights (1 MB)
                const float* __restrict__ bp,          // packed bias [1024]
                const float* __restrict__ c0,          // initial c for this layer [256]
                unsigned short* __restrict__ hbuf,     // state/exchange [(T+1),B,H] bf16
                unsigned int* __restrict__ flagsP)     // [NGROUP][T]
{
  __shared__ __align__(1024) unsigned char sW[128 * 1024];
  __shared__ __align__(1024) unsigned char sA[32 * 1024];

  const unsigned tid  = threadIdx.x;
  const unsigned wave = tid >> 5;
  const unsigned lane = tid & 31u;
  const unsigned s    = blockIdx.x & 7u;     // unit slice
  const unsigned gb   = blockIdx.x >> 3;     // batch group
  const unsigned b0row = gb * 32u;

  // ---- one-time: pull the 128 KB resident weight slice into LDS ----
#if USE_TDM
  if (wave == 0) {
    unsigned lds_addr = (unsigned)(unsigned long long)(void*)sW;
    unsigned long long ga =
        (unsigned long long)(const void*)Wp + (unsigned long long)s * 131072ull;
    u32x4v g0;
    g0[0] = 1u;                                   // count=1, user descriptor
    g0[1] = lds_addr;                             // lds_addr
    g0[2] = (unsigned)ga;                         // global_addr[31:0]
    g0[3] = (unsigned)((ga >> 32) & 0x01FFFFFFull) | (2u << 30);  // addr hi | type=2
    i32x8v g1;
    g1[0] = (int)(2u << 16);                      // data_size=2 (4B units)
    g1[1] = (int)0x80000000u;                     // tensor_dim0[15:0]=32768 dwords
    g1[2] = (int)(1u << 16);                      // tensor_dim1=1
    g1[3] = (int)0x80000000u;                     // tile_dim0=32768 dwords (128 KB)
    g1[4] = 0;                                    // tile_dim1/2 unused (1-D)
    g1[5] = 32768;                                // tensor_dim0_stride
    g1[6] = 0;
    g1[7] = 0;
    i32x4v g2 = {0, 0, 0, 0};
    i32x4v g3 = {0, 0, 0, 0};
    i32x8v g4 = {0, 0, 0, 0, 0, 0, 0, 0};         // trailing group (clang-23 6-arg form)
    __builtin_amdgcn_tensor_load_to_lds(g0, g1, g2, g3, g4, 0);
    __builtin_amdgcn_s_wait_tensorcnt(0);
  }
#else
  {
    const uint4* src = (const uint4*)(Wp + (size_t)s * 65536u);
    uint4* dst = (uint4*)sW;
#pragma unroll
    for (int i = 0; i < 32; ++i) dst[i * 256 + tid] = src[i * 256 + tid];
  }
#endif
  __syncthreads();

  // per-wave WMMA tile assignment: mt in {0,1}, two local n-tiles
  const unsigned mt   = wave & 1u;
  const unsigned ntl0 = (wave >> 1) << 1;
  const float bias0 = bp[s * 128u + ntl0 * 16u + (lane & 15u)];
  const float bias1 = bp[s * 128u + (ntl0 + 1u) * 16u + (lane & 15u)];

  // per-thread cell-state ownership: row rr (0..31), 4 units at uu
  const unsigned rr = tid >> 3;
  const unsigned uu = (tid & 7u) << 2;
  float c_[4];
#pragma unroll
  for (int k = 0; k < 4; ++k) c_[k] = c0[s * 32u + uu + k];

  float* gates = (float*)sA;  // aliases the A staging region after the WMMA pass

  for (int t = 0; t < T_; ++t) {
    // ---- wait for full h(t) from all 8 slices of this batch group ----
    if (t > 0) {
      if (tid == 0) {
        while (__hip_atomic_load(&flagsP[gb * T_ + (t - 1)], __ATOMIC_ACQUIRE,
                                 __HIP_MEMORY_SCOPE_AGENT) < NSLICE)
          __builtin_amdgcn_s_sleep(1);
      }
      __syncthreads();
      __threadfence();
    }

    // ---- stage A (bf16, fragment order): low K = x_t, high K = h(t) ----
    if (is_f32) {
#pragma unroll
      for (int it = 0; it < 8; ++it) {
        unsigned flat = (unsigned)it * 256u + tid;
        unsigned r = flat >> 6, kq = (flat & 63u) << 2;
        const float* srcp = xf + (((size_t)(b0row + r) * T_ + (unsigned)t) * C_ + kq);
        if (t + 1 < T_) __builtin_prefetch(srcp + C_, 0, 1);   // next timestep row
        const float4 xv = *(const float4*)srcp;
        unsigned d0 = (unsigned)f2bf(xv.x) | ((unsigned)f2bf(xv.y) << 16);
        unsigned d1 = (unsigned)f2bf(xv.z) | ((unsigned)f2bf(xv.w) << 16);
        unsigned kt = kq >> 5, kk = kq & 31u;
        unsigned ln = (r & 15u) + (((kk & 15u) >= 8u) ? 16u : 0u);
        unsigned v = ((kk >> 4) << 2) + ((kk & 7u) >> 1);
        unsigned off = (((r >> 4) * 16u + kt) << 10) + ln * 32u + v * 4u;
        *(uint2*)(sA + off) = make_uint2(d0, d1);
      }
    } else {
#pragma unroll
      for (int it = 0; it < 8; ++it) {
        unsigned flat = (unsigned)it * 256u + tid;
        unsigned r = flat >> 6, kq = (flat & 63u) << 2;
        const unsigned short* srcp =
            xb + ((size_t)(t + 1) * B_ + (b0row + r)) * H_ + kq;
        if (t + 1 < T_) __builtin_prefetch(srcp + (size_t)B_ * H_, 0, 1);
        const uint2 hv = *(const uint2*)srcp;     // 4 bf16, already pair-packed
        unsigned kt = kq >> 5, kk = kq & 31u;
        unsigned ln = (r & 15u) + (((kk & 15u) >= 8u) ? 16u : 0u);
        unsigned v = ((kk >> 4) << 2) + ((kk & 7u) >> 1);
        unsigned off = (((r >> 4) * 16u + kt) << 10) + ln * 32u + v * 4u;
        *(uint2*)(sA + off) = hv;
      }
    }
#pragma unroll
    for (int it = 0; it < 8; ++it) {              // high K half: h(t) from own hbuf
      unsigned flat = (unsigned)it * 256u + tid;
      unsigned r = flat >> 6, jq = (flat & 63u) << 2;
      const uint2 hv =
          *(const uint2*)(hbuf + ((size_t)t * B_ + (b0row + r)) * H_ + jq);
      unsigned k = 256u + jq;
      unsigned kt = k >> 5, kk = k & 31u;
      unsigned ln = (r & 15u) + (((kk & 15u) >= 8u) ? 16u : 0u);
      unsigned v = ((kk >> 4) << 2) + ((kk & 7u) >> 1);
      unsigned off = (((r >> 4) * 16u + kt) << 10) + ln * 32u + v * 4u;
      *(uint2*)(sA + off) = hv;
    }
    __syncthreads();

    // ---- WMMA: gates slice = A[32x512] x Wslice[512x32(xN2)] + bias ----
    v8f acc0 = {bias0, bias0, bias0, bias0, bias0, bias0, bias0, bias0};
    v8f acc1 = {bias1, bias1, bias1, bias1, bias1, bias1, bias1, bias1};
#pragma unroll
    for (int kt = 0; kt < 16; ++kt) {
      v16bf a  = *(const v16bf*)(sA + ((mt * 16u + kt) << 10) + lane * 32u);
      v16bf w0 = *(const v16bf*)(sW + ((ntl0 * 16u + kt) << 10) + lane * 32u);
      acc0 = __builtin_amdgcn_wmma_f32_16x16x32_bf16(false, a, false, w0,
                                                     (short)0, acc0, false, false);
      v16bf w1 = *(const v16bf*)(sW + (((ntl0 + 1u) * 16u + kt) << 10) + lane * 32u);
      acc1 = __builtin_amdgcn_wmma_f32_16x16x32_bf16(false, a, false, w1,
                                                     (short)0, acc1, false, false);
    }
    __syncthreads();   // everyone done reading sA; safe to overwrite as gates

    // ---- dump accumulators (C-layout) into LDS gates buffer [32][128] ----
    {
      float a0[8] __attribute__((aligned(32)));
      float a1[8] __attribute__((aligned(32)));
      *(v8f*)a0 = acc0;
      *(v8f*)a1 = acc1;
      unsigned rbase = mt * 16u + ((lane >> 4) << 3);
      unsigned cb = lane & 15u;
#pragma unroll
      for (int q = 0; q < 8; ++q) {
        gates[(rbase + q) * 128u + ntl0 * 16u + cb]        = a0[q];
        gates[(rbase + q) * 128u + (ntl0 + 1u) * 16u + cb] = a1[q];
      }
    }
    __syncthreads();

    // ---- elementwise LSTM cell update; h_new -> exchange buffer ----
    {
      const float4 iv = *(const float4*)(gates + rr * 128u + 0u  + uu);
      const float4 fv = *(const float4*)(gates + rr * 128u + 32u + uu);
      const float4 gv = *(const float4*)(gates + rr * 128u + 64u + uu);
      const float4 ov = *(const float4*)(gates + rr * 128u + 96u + uu);
      float ia[4] = {iv.x, iv.y, iv.z, iv.w};
      float fa[4] = {fv.x, fv.y, fv.z, fv.w};
      float ga[4] = {gv.x, gv.y, gv.z, gv.w};
      float oa[4] = {ov.x, ov.y, ov.z, ov.w};
      float hn[4];
#pragma unroll
      for (int k = 0; k < 4; ++k) {
        float ig = sigm(ia[k]);
        float fg = sigm(fa[k]);
        float gg = tanh_fast(ga[k]);
        float og = sigm(oa[k]);
        c_[k] = fg * c_[k] + ig * gg;
        hn[k] = og * tanh_fast(c_[k]);
      }
      unsigned d0 = (unsigned)f2bf(hn[0]) | ((unsigned)f2bf(hn[1]) << 16);
      unsigned d1 = (unsigned)f2bf(hn[2]) | ((unsigned)f2bf(hn[3]) << 16);
      *(uint2*)(hbuf + ((size_t)(t + 1) * B_ + (b0row + rr)) * H_ +
                (s * 32u + uu)) = make_uint2(d0, d1);
    }
    __threadfence();
    __syncthreads();
    if (tid == 0)
      __hip_atomic_fetch_add(&flagsP[gb * T_ + t], 1u, __ATOMIC_RELEASE,
                             __HIP_MEMORY_SCOPE_AGENT);
  }
}

// ---------------------------------------------------------------------------
// Head: out[b,t] = dot(h2[t,b,:], Wd) + bd
// ---------------------------------------------------------------------------
__global__ void lstm_proj(const unsigned short* __restrict__ hb1,
                          const float* __restrict__ Wd,
                          const float* __restrict__ bd,
                          float* __restrict__ out) {
  unsigned idx = blockIdx.x * blockDim.x + threadIdx.x;  // B*T
  unsigned b = idx >> 9, t = idx & 511u;
  const unsigned short* row = hb1 + ((size_t)(t + 1) * B_ + b) * H_;
  float acc = bd[0];
#pragma unroll 8
  for (int j = 0; j < H_; ++j) acc = fmaf(bf2f(row[j]), Wd[j], acc);
  out[idx] = acc;
}

// ---------------------------------------------------------------------------
extern "C" void kernel_launch(void* const* d_in, const int* in_sizes, int n_in,
                              void* d_out, int out_size, void* d_ws, size_t ws_size,
                              hipStream_t stream) {
  (void)in_sizes; (void)n_in; (void)out_size; (void)ws_size;
  const float* x    = (const float*)d_in[0];
  const float* h0   = (const float*)d_in[1];
  const float* c0   = (const float*)d_in[2];
  const float* Wih0 = (const float*)d_in[3];
  const float* Whh0 = (const float*)d_in[4];
  const float* bih0 = (const float*)d_in[5];
  const float* bhh0 = (const float*)d_in[6];
  const float* Wih1 = (const float*)d_in[7];
  const float* Whh1 = (const float*)d_in[8];
  const float* bih1 = (const float*)d_in[9];
  const float* bhh1 = (const float*)d_in[10];
  const float* Wd   = (const float*)d_in[11];
  const float* bd   = (const float*)d_in[12];

  char* ws = (char*)d_ws;
  constexpr size_t OFF_W0    = 0;
  constexpr size_t OFF_W1    = (size_t)1 << 20;
  constexpr size_t OFF_B0    = (size_t)2 << 20;
  constexpr size_t OFF_B1    = OFF_B0 + 4096;
  constexpr size_t OFF_FLAGS = OFF_B0 + 8192;
  constexpr size_t OFF_H0    = (size_t)4 << 20;
  constexpr size_t HBYTES    = (size_t)(T_ + 1) * B_ * H_ * 2;   // ~64.1 MB
  constexpr size_t OFF_H1    = OFF_H0 + ((HBYTES + ((size_t)1 << 20) - 1) &
                                         ~(((size_t)1 << 20) - 1));

  unsigned short* W0p = (unsigned short*)(ws + OFF_W0);
  unsigned short* W1p = (unsigned short*)(ws + OFF_W1);
  float* b0p = (float*)(ws + OFF_B0);
  float* b1p = (float*)(ws + OFF_B1);
  unsigned int* flags = (unsigned int*)(ws + OFF_FLAGS);
  unsigned short* hb0 = (unsigned short*)(ws + OFF_H0);
  unsigned short* hb1 = (unsigned short*)(ws + OFF_H1);

  lstm_pack_weights<<<4096, 256, 0, stream>>>(Wih0, Whh0, Wih1, Whh1, W0p, W1p);
  lstm_init<<<512, 256, 0, stream>>>(h0, bih0, bhh0, bih1, bhh1, flags, hb0, hb1,
                                     b0p, b1p);
  lstm_phase<<<NGROUP * NSLICE, 256, 0, stream>>>(x, nullptr, 1, W0p, b0p,
                                                  c0 + 0, hb0, flags + 0);
  lstm_phase<<<NGROUP * NSLICE, 256, 0, stream>>>(nullptr, hb0, 0, W1p, b1p,
                                                  c0 + H_, hb1,
                                                  flags + NGROUP * T_);
  lstm_proj<<<512, 256, 0, stream>>>(hb1, Wd, bd, (float*)d_out);
}
